// WindowAttention_26147760898426
// MI455X (gfx1250) — matrix operand order repaired and checked by reading, not compiled, not run on hardware
//
#include <hip/hip_runtime.h>

// ---------------------------------------------------------------------------
// Window attention for MI455X (gfx1250): bf16 WMMA everywhere, f32 softmax.
// One workgroup (128 threads = 4 waves, wave32) per window.
// ---------------------------------------------------------------------------

typedef __attribute__((ext_vector_type(16))) __bf16 bf16x16;
typedef __attribute__((ext_vector_type(8)))  float  f32x8;

#define DIM 128
#define NHEAD 4
#define HD 32
#define NTOK 64
#define QKV_STRIDE 256   // Q cols 0..127, K cols 128..255 (bf16)
#define SCALE 0.17677669529663687f  // 1/sqrt(32)

// ws layout (bytes)
#define WS_QKV_OFF  0         // 384*128 bf16 = 98304 B
#define WS_PROJ_OFF 98304     // 128*128 bf16 = 32768 B
#define WS_BIAS_OFF 131072    // 4*64*64 f32  = 65536 B

__device__ __forceinline__ unsigned short f2bf_bits(float f) {
    union { float f; unsigned int u; } cv; cv.f = f;
    unsigned int u = cv.u;
    u += 0x7fffu + ((u >> 16) & 1u);   // round-to-nearest-even
    return (unsigned short)(u >> 16);
}

// A-matrix 16x32 bf16 fragment (M x K), row-major source, K contiguous.
// lane(0-15): M=lane, K base 0 ; lane(16-31): M=lane-16, K base 8.
// vector elems 0-7 -> K=base..base+7 ; elems 8-15 -> K=base+16..base+23.
__device__ __forceinline__ bf16x16 load_a_frag(const unsigned short* base,
                                               int stride, int mbase, int kbase,
                                               int lane) {
    int l15 = lane & 15, lh = lane >> 4;
    const unsigned short* p = base + (mbase + l15) * stride + kbase + lh * 8;
    union { bf16x16 v; uint4 q[2]; } u;
    u.q[0] = *(const uint4*)(p);
    u.q[1] = *(const uint4*)(p + 16);
    return u.v;
}

// B-matrix 32x16 bf16 fragment (K x N), source holds B[n][k] with k contiguous
// (i.e. the N-major / "needs K contiguous per row" layout we stage in LDS/ws).
// lane(0-15): N=lane, K base 0 ; lane(16-31): N=lane-16, K base 8; elems 0-7 -> K=base..base+7, elems 8-15 -> K=base+16..base+23.
__device__ __forceinline__ bf16x16 load_b_frag(const unsigned short* base,
                                               int stride, int nbase, int kbase,
                                               int lane) {
    int l15 = lane & 15, lh = lane >> 4;
    const unsigned short* p = base + (nbase + l15) * stride + kbase + lh * 8;
    union { bf16x16 v; uint4 q[2]; } u;
    u.q[0] = *(const uint4*)(p);
    u.q[1] = *(const uint4*)(p + 16);
    return u.v;
}

__device__ __forceinline__ f32x8 wmma_bf16(bf16x16 a, bf16x16 b, f32x8 c) {
    return __builtin_amdgcn_wmma_f32_16x16x32_bf16(
        /*neg_a=*/false, a, /*neg_b=*/false, b,
        /*c_mod=*/(short)0, c, /*reuse_a=*/false, /*reuse_b=*/false);
}

// ---------------------------------------------------------------------------
// Pre-kernel: convert weights to bf16, gather relative-position bias.
// ---------------------------------------------------------------------------
__global__ void prep_kernel(const float* __restrict__ w_qkv,
                            const float* __restrict__ w_proj,
                            const float* __restrict__ bias_table,
                            const int*   __restrict__ rel_idx,
                            unsigned short* __restrict__ ws_qkv,
                            unsigned short* __restrict__ ws_proj,
                            float* __restrict__ ws_bias) {
    int i = blockIdx.x * blockDim.x + threadIdx.x;
    if (i < 384 * 128) ws_qkv[i] = f2bf_bits(w_qkv[i]);
    if (i < 128 * 128) ws_proj[i] = f2bf_bits(w_proj[i]);
    if (i < 64 * 64) {
        int t = rel_idx[i];
        #pragma unroll
        for (int h = 0; h < NHEAD; ++h)
            ws_bias[h * 4096 + i] = bias_table[t * NHEAD + h];
    }
}

// ---------------------------------------------------------------------------
// Main kernel: one block per window.
// Dynamic LDS (128 KB):
//   [0,      16384)  Xbf  bf16[64][128]   (later reused as merged O)
//   [16384,  49152)  QKVbf bf16[64][256]  (Q | K, per-token rows)
//   [49152,  65536)  Vt   bf16[4][32][64] (V transposed, per head)
//   [65536, 131072)  S    f32 [4][64][64] (scores; P bf16 overlays rows)
// ---------------------------------------------------------------------------
__global__ void __launch_bounds__(128)
win_attn_kernel(const float* __restrict__ x,
                const unsigned short* __restrict__ Wqkv,   // [384][128] bf16
                const unsigned short* __restrict__ Wproj,  // [128][128] bf16
                const float* __restrict__ biasg,           // [4][64][64]
                const float* __restrict__ bproj,           // [128]
                float* __restrict__ out) {
    extern __shared__ char smem[];
    unsigned short* Xbf   = (unsigned short*)(smem);
    unsigned short* QKVbf = (unsigned short*)(smem + 16384);
    unsigned short* Vt    = (unsigned short*)(smem + 49152);
    float*          S     = (float*)(smem + 65536);

    const int tid  = threadIdx.x;
    const int lane = tid & 31;
    const int wave = tid >> 5;
    const int l15  = lane & 15;
    const int lh   = lane >> 4;

    // ---- load x tile, convert to bf16 in LDS -----------------------------
    const float4* xg = (const float4*)(x + (size_t)blockIdx.x * (NTOK * DIM));
    for (int i = tid; i < (NTOK * DIM) / 4; i += 128) {
        float4 t = xg[i];
        uint2 pk;
        pk.x = (unsigned)f2bf_bits(t.x) | ((unsigned)f2bf_bits(t.y) << 16);
        pk.y = (unsigned)f2bf_bits(t.z) | ((unsigned)f2bf_bits(t.w) << 16);
        ((uint2*)Xbf)[i] = pk;
    }
    __syncthreads();

    // ---- stage 1: QKV GEMM  [64,128] x [128,384] -------------------------
    // wave owns 6 N-tiles (96 output cols), all 4 M-tiles, 4 k-steps.
    for (int nti = 0; nti < 6; ++nti) {
        const int ntile = wave * 6 + nti;
        const int nbase = ntile * 16;
        f32x8 acc[4];
        #pragma unroll
        for (int mt = 0; mt < 4; ++mt)
            #pragma unroll
            for (int r = 0; r < 8; ++r) acc[mt][r] = 0.0f;

        #pragma unroll
        for (int ks = 0; ks < 4; ++ks) {
            bf16x16 b = load_b_frag(Wqkv, DIM, nbase, ks * 32, lane);
            #pragma unroll
            for (int mt = 0; mt < 4; ++mt) {
                bf16x16 a = load_a_frag(Xbf, DIM, mt * 16, ks * 32, lane);
                acc[mt] = wmma_bf16(a, b, acc[mt]);
            }
        }

        const int part = ntile >> 3;   // 0=Q 1=K 2=V
        #pragma unroll
        for (int mt = 0; mt < 4; ++mt) {
            #pragma unroll
            for (int r = 0; r < 8; ++r) {
                const int token = mt * 16 + r + 8 * lh;
                float v = acc[mt][r];
                if (part == 0) {
                    QKVbf[token * QKV_STRIDE + nbase + l15] = f2bf_bits(v * SCALE);
                } else if (part == 1) {
                    QKVbf[token * QKV_STRIDE + nbase + l15] = f2bf_bits(v);
                } else {
                    const int head = (ntile - 16) >> 1;
                    const int dl   = ((ntile - 16) & 1) * 16 + l15;
                    Vt[head * 2048 + dl * 64 + token] = f2bf_bits(v);
                }
            }
        }
    }
    __syncthreads();

    // ---- stage 2: S = (Q*scale) K^T + bias   (wave = head) ---------------
    const int h = wave;
    float* Sh = S + h * 4096;
    const float* bh = biasg + h * 4096;
    #pragma unroll
    for (int mt = 0; mt < 4; ++mt) {
        bf16x16 a = load_a_frag(QKVbf, QKV_STRIDE, mt * 16, h * HD, lane);
        #pragma unroll
        for (int nt = 0; nt < 4; ++nt) {
            bf16x16 b = load_b_frag(QKVbf, QKV_STRIDE, nt * 16, 128 + h * HD, lane);
            f32x8 c;
            #pragma unroll
            for (int r = 0; r < 8; ++r)
                c[r] = bh[(mt * 16 + r + 8 * lh) * 64 + nt * 16 + l15];
            c = wmma_bf16(a, b, c);
            #pragma unroll
            for (int r = 0; r < 8; ++r)
                Sh[(mt * 16 + r + 8 * lh) * 64 + nt * 16 + l15] = c[r];
        }
    }
    __syncthreads();

    // ---- softmax: 2 rows per lane; P (bf16) overlays S rows --------------
    for (int rr = 0; rr < 2; ++rr) {
        const int row = rr * 32 + lane;
        const float4* Sr = (const float4*)(Sh + row * 64);
        float4 vb[16];
        float mx = -3.402823466e38f;
        #pragma unroll
        for (int i = 0; i < 16; ++i) {
            vb[i] = Sr[i];
            mx = fmaxf(mx, fmaxf(fmaxf(vb[i].x, vb[i].y), fmaxf(vb[i].z, vb[i].w)));
        }
        float sum = 0.0f;
        #pragma unroll
        for (int i = 0; i < 16; ++i) {
            vb[i].x = __expf(vb[i].x - mx);
            vb[i].y = __expf(vb[i].y - mx);
            vb[i].z = __expf(vb[i].z - mx);
            vb[i].w = __expf(vb[i].w - mx);
            sum += vb[i].x + vb[i].y + vb[i].z + vb[i].w;
        }
        const float inv = 1.0f / sum;
        uint2* Pr = (uint2*)(Sh + row * 64);   // same row byte offset: safe overlay
        #pragma unroll
        for (int i = 0; i < 16; ++i) {
            uint2 pk;
            pk.x = (unsigned)f2bf_bits(vb[i].x * inv) | ((unsigned)f2bf_bits(vb[i].y * inv) << 16);
            pk.y = (unsigned)f2bf_bits(vb[i].z * inv) | ((unsigned)f2bf_bits(vb[i].w * inv) << 16);
            Pr[i] = pk;
        }
    }
    __syncthreads();

    // ---- stage 3: O = P V   (P row stride = 128 bf16 elems) --------------
    unsigned short* Pb = (unsigned short*)Sh;      // [64 rows] stride 128
    unsigned short* Vh = Vt + h * 2048;            // [32][64]
    #pragma unroll
    for (int mt = 0; mt < 4; ++mt) {
        #pragma unroll
        for (int nt = 0; nt < 2; ++nt) {
            f32x8 c;
            #pragma unroll
            for (int r = 0; r < 8; ++r) c[r] = 0.0f;
            #pragma unroll
            for (int ks = 0; ks < 2; ++ks) {
                bf16x16 a = load_a_frag(Pb, 128, mt * 16, ks * 32, lane);
                bf16x16 b = load_b_frag(Vh, 64, nt * 16, ks * 32, lane);
                c = wmma_bf16(a, b, c);
            }
            #pragma unroll
            for (int r = 0; r < 8; ++r) {
                const int token = mt * 16 + r + 8 * lh;
                Xbf[token * DIM + h * HD + nt * 16 + l15] = f2bf_bits(c[r]); // merged O
            }
        }
    }
    __syncthreads();

    // ---- stage 4: Y = O Wproj^T + b  -------------------------------------
    float* outw = out + (size_t)blockIdx.x * (NTOK * DIM);
    for (int nti = 0; nti < 2; ++nti) {
        const int nbase = (wave * 2 + nti) * 16;
        const float bp = bproj[nbase + l15];
        f32x8 acc[4];
        #pragma unroll
        for (int mt = 0; mt < 4; ++mt)
            #pragma unroll
            for (int r = 0; r < 8; ++r) acc[mt][r] = 0.0f;

        #pragma unroll
        for (int ks = 0; ks < 4; ++ks) {
            bf16x16 b = load_b_frag(Wproj, DIM, nbase, ks * 32, lane);
            #pragma unroll
            for (int mt = 0; mt < 4; ++mt) {
                bf16x16 a = load_a_frag(Xbf, DIM, mt * 16, ks * 32, lane);
                acc[mt] = wmma_bf16(a, b, acc[mt]);
            }
        }
        #pragma unroll
        for (int mt = 0; mt < 4; ++mt)
            #pragma unroll
            for (int r = 0; r < 8; ++r)
                outw[(mt * 16 + r + 8 * lh) * DIM + nbase + l15] = acc[mt][r] + bp;
    }
}

// ---------------------------------------------------------------------------
extern "C" void kernel_launch(void* const* d_in, const int* in_sizes, int n_in,
                              void* d_out, int out_size, void* d_ws, size_t ws_size,
                              hipStream_t stream) {
    const float* x          = (const float*)d_in[0];
    const float* w_qkv      = (const float*)d_in[1];
    const float* w_proj     = (const float*)d_in[2];
    const float* b_proj     = (const float*)d_in[3];
    const float* bias_table = (const float*)d_in[4];
    const int*   rel_idx    = (const int*)d_in[5];

    unsigned short* ws_qkv  = (unsigned short*)((char*)d_ws + WS_QKV_OFF);
    unsigned short* ws_proj = (unsigned short*)((char*)d_ws + WS_PROJ_OFF);
    float*          ws_bias = (float*)((char*)d_ws + WS_BIAS_OFF);

    prep_kernel<<<192, 256, 0, stream>>>(w_qkv, w_proj, bias_table, rel_idx,
                                         ws_qkv, ws_proj, ws_bias);

    win_attn_kernel<<<8192, 128, 131072, stream>>>(x, ws_qkv, ws_proj, ws_bias,
                                                   b_proj, (float*)d_out);
}
